// MTR_15350213116590
// MI455X (gfx1250) — matrix operand order repaired and checked by reading, not compiled, not run on hardware
//
#include <hip/hip_runtime.h>
#include <hip/hip_bf16.h>
#include <math.h>

typedef __attribute__((ext_vector_type(16))) __bf16 v16bf;
typedef __attribute__((ext_vector_type(8)))  __bf16 v8bf;
typedef __attribute__((ext_vector_type(8)))  float  v8f;

#define T_STEPS 32
#define DMODEL  128
#define E_TOT   100
#define E_PAD   112

// ---- WMMA fragment helpers (wave32 layouts per CDNA5 ISA 7.12.2) ----
// A matrix 16x32 bf16: lane (0..15) row M=lane, K chunks {hi*8..} and {16+hi*8..}
__device__ __forceinline__ v16bf ldA(const __bf16* base, int row, int stride, int kbase, int hi) {
  union { v16bf v; v8bf h[2]; } u;
  const __bf16* p = base + row * stride + kbase + hi * 8;
  u.h[0] = *(const v8bf*)(p);
  u.h[1] = *(const v8bf*)(p + 16);
  return u.v;
}
// B matrix 32x16 bf16: lane holds one column, K chunks {hi*8..} and {16+hi*8..}
__device__ __forceinline__ v16bf ldB(const __bf16* base, int col, int stride, int kbase, int hi) {
  union { v16bf v; v8bf h[2]; } u;
  const __bf16* p = base + col * stride + kbase + hi * 8;
  u.h[0] = *(const v8bf*)(p);
  u.h[1] = *(const v8bf*)(p + 16);
  return u.v;
}
__device__ __forceinline__ v8f wmma_bf16(v16bf a, v16bf b, v8f c) {
  return __builtin_amdgcn_wmma_f32_16x16x32_bf16(false, a, false, b, (short)0, c, false, false);
}
__device__ __forceinline__ __bf16 tobf(float f) { return (__bf16)f; }

#define ZERO8 {0.f,0.f,0.f,0.f,0.f,0.f,0.f,0.f}

// =====================================================================
// Subnet helpers
// =====================================================================
__device__ __forceinline__ void stage_w(const float* __restrict__ W, const float* __restrict__ B,
                                        int Kin, int Kdim, __bf16* Wt, float* bias_s, int tid) {
  for (int i = tid; i < 64 * Kdim; i += 256) {
    int n = i / Kdim, k = i % Kdim;
    Wt[n * 128 + k] = tobf((k < Kin) ? W[k * 64 + n] : 0.f);
  }
  if (tid < 64) bias_s[tid] = B[tid];
}

// GEMM: Y[N x 64] = relu(X[N x (KSTEPS*32)] @ W + b); fully unrolled, fixed regs.
template<int KSTEPS, int MTILES>
__device__ __forceinline__ void subnet_gemm(const __bf16* X, const __bf16* Wt,
                                            const float* bias_s, __bf16* Y,
                                            int wave, int lr, int hi) {
  #pragma unroll
  for (int mi = 0; mi < MTILES / 8; ++mi) {
    const int mt = wave + mi * 8;
    v16bf afr[KSTEPS];
    #pragma unroll
    for (int ks = 0; ks < KSTEPS; ++ks)
      afr[ks] = ldA(X, mt * 16 + lr, 128, ks * 32, hi);
    #pragma unroll
    for (int nt = 0; nt < 4; ++nt) {
      v8f acc = ZERO8;
      #pragma unroll
      for (int ks = 0; ks < KSTEPS; ++ks)
        acc = wmma_bf16(afr[ks], ldB(Wt, nt * 16 + lr, 128, ks * 32, hi), acc);
      const int n = nt * 16 + lr;
      const float bv = bias_s[n];
      #pragma unroll
      for (int r = 0; r < 8; ++r) {
        const int m = mt * 16 + r + hi * 8;
        float v = acc[r] + bv;
        Y[m * 128 + n] = tobf(v > 0.f ? v : 0.f);
      }
    }
  }
}

// column-wise max over N node rows of Y (cols 0..63) -> agg_s[64]
template<int N>
__device__ __forceinline__ void col_max(const __bf16* Y, float* red, float* agg_s, int tid) {
  const int col = tid & 63, part = tid >> 6;
  const int chunk = N >> 2;
  float m = -3.0e38f;
  for (int r = part * chunk; r < (part + 1) * chunk; ++r)
    m = fmaxf(m, (float)Y[r * 128 + col]);
  red[part * 64 + col] = m;
  __syncthreads();
  if (tid < 64)
    agg_s[tid] = fmaxf(fmaxf(red[tid], red[64 + tid]),
                       fmaxf(red[128 + tid], red[192 + tid]));
  __syncthreads();
}

// =====================================================================
// Subnet: 3x (Linear -> ReLU -> maxpool concat), final max over nodes.
// One block per (graph, t). Templated on node count N (256 or 128).
// =====================================================================
template<int N>
__global__ __launch_bounds__(256)
void subnet_kernel(const float* __restrict__ feat,
                   const float* __restrict__ W0, const float* __restrict__ B0,
                   const float* __restrict__ W1, const float* __restrict__ B1,
                   const float* __restrict__ W2, const float* __restrict__ B2,
                   float* __restrict__ Gbuf, int EOFF, int nanmask)
{
  __shared__ __bf16 bufA[256 * 128];   // 64 KB
  __shared__ __bf16 bufB[256 * 128];   // 64 KB
  __shared__ __bf16 Wt[64 * 128];      // 16 KB (W^T, stride 128)
  __shared__ float  bias_s[64];
  __shared__ float  red[4 * 64];
  __shared__ float  agg_s[64];

  const int tid  = threadIdx.x;
  const int g    = blockIdx.x / T_STEPS;
  const int t    = blockIdx.x % T_STEPS;
  const int wave = tid >> 5;
  const int lane = tid & 31;
  const int lr   = lane & 15;
  const int hi   = lane >> 4;
  constexpr int MT = N >> 4;

  // stage layer-0 input: fp32 -> bf16, K padded 16 -> 32 (zeros)
  const float* src0 = feat + (size_t)(g * T_STEPS + t) * N * 16;
  for (int i = tid; i < N * 32; i += 256) {
    int row = i >> 5, col = i & 31;
    float v = 0.f;
    if (col < 16) {
      v = src0[row * 16 + col];
      if (nanmask && !(v == v)) v = 0.f;
    }
    bufA[row * 128 + col] = tobf(v);
  }
  stage_w(W0, B0, 16, 32, Wt, bias_s, tid);
  __syncthreads();

  // ---- layer 0: bufA(K=32 padded) -> bufB ----
  subnet_gemm<1, MT>(bufA, Wt, bias_s, bufB, wave, lr, hi);
  __syncthreads();
  col_max<N>(bufB, red, agg_s, tid);
  for (int i = tid; i < N * 64; i += 256) {
    int r = i >> 6, c = i & 63;
    bufB[r * 128 + 64 + c] = tobf(agg_s[c]);
  }
  stage_w(W1, B1, 128, 128, Wt, bias_s, tid);   // Wt was consumed before col_max sync
  __syncthreads();

  // ---- layer 1: bufB -> bufA ----
  subnet_gemm<4, MT>(bufB, Wt, bias_s, bufA, wave, lr, hi);
  __syncthreads();
  col_max<N>(bufA, red, agg_s, tid);
  for (int i = tid; i < N * 64; i += 256) {
    int r = i >> 6, c = i & 63;
    bufA[r * 128 + 64 + c] = tobf(agg_s[c]);
  }
  stage_w(W2, B2, 128, 128, Wt, bias_s, tid);
  __syncthreads();

  // ---- layer 2: bufA -> bufB ----
  subnet_gemm<4, MT>(bufA, Wt, bias_s, bufB, wave, lr, hi);
  __syncthreads();
  col_max<N>(bufB, red, agg_s, tid);

  // final feature = max_nodes(concat(h, agg)) = [agg, agg]
  if (tid < 128)
    Gbuf[((size_t)t * E_TOT + EOFF + g) * DMODEL + tid] = agg_s[tid & 63];
}

// =====================================================================
// Self-attention per timestep. E=100 padded to 112, D=128.
// LDS arena (148 KB) with phase-based aliasing.
// =====================================================================
#define ARENA_BYTES 151552
__global__ __launch_bounds__(256)
void attn_kernel(const float* __restrict__ Gbuf,
                 const float* __restrict__ Wq, const float* __restrict__ Wk,
                 const float* __restrict__ Wv, float* __restrict__ obuf)
{
  __shared__ __align__(16) char arena[ARENA_BYTES];
  __bf16* xb = (__bf16*)(arena);            // [112][128] bf16  28672 B  (phase 1)
  __bf16* Wt = (__bf16*)(arena + 28672);    // [128][128] bf16  32768 B  (phase 1)
  float*  sc = (float*) (arena);            // [112][112] f32   50176 B  (aliases xb+Wt)
  __bf16* qb = (__bf16*)(arena + 61440);    // [112][128] bf16
  __bf16* at = qb;                          // attn [112][128]  (aliases qb after scores)
  __bf16* kb = (__bf16*)(arena + 90112);    // [112][128] bf16
  __bf16* vt = (__bf16*)(arena + 118784);   // [128][128] bf16 (v^T, K padded)

  const int tid  = threadIdx.x;
  const int t    = blockIdx.x;
  const int wave = tid >> 5, lane = tid & 31, lr = lane & 15, hi = lane >> 4;

  for (int i = tid; i < E_PAD * DMODEL; i += 256) {
    int e = i >> 7, d = i & 127;
    xb[i] = tobf((e < E_TOT) ? Gbuf[((size_t)t * E_TOT + e) * DMODEL + d] : 0.f);
  }
  for (int i = tid; i < DMODEL * DMODEL; i += 256) vt[i] = tobf(0.f);

  const float* Wsrc[3] = {Wq, Wk, Wv};
  #pragma unroll
  for (int p = 0; p < 3; ++p) {
    __syncthreads();
    for (int i = tid; i < DMODEL * DMODEL; i += 256) {
      int n = i >> 7, k = i & 127;
      Wt[n * 128 + k] = tobf(Wsrc[p][k * 128 + n]);
    }
    __syncthreads();
    for (int tile = wave; tile < 7 * 8; tile += 8) {
      int mt = tile >> 3, nt = tile & 7;
      v8f acc = ZERO8;
      #pragma unroll
      for (int ks = 0; ks < 4; ++ks) {
        v16bf a = ldA(xb, mt * 16 + lr, 128, ks * 32, hi);
        v16bf b = ldB(Wt, nt * 16 + lr, 128, ks * 32, hi);
        acc = wmma_bf16(a, b, acc);
      }
      int n = nt * 16 + lr;
      #pragma unroll
      for (int r = 0; r < 8; ++r) {
        int m = mt * 16 + r + hi * 8;
        if (p == 0)      qb[m * 128 + n] = tobf(acc[r]);
        else if (p == 1) kb[m * 128 + n] = tobf(acc[r]);
        else             vt[n * 128 + m] = tobf(acc[r]);   // transposed store
      }
    }
  }
  __syncthreads();

  // scores = q @ k^T  (B = k rows directly: [n=e][k=d])
  for (int tile = wave; tile < 7 * 7; tile += 8) {
    int mt = tile / 7, nt = tile % 7;
    v8f acc = ZERO8;
    #pragma unroll
    for (int ks = 0; ks < 4; ++ks) {
      v16bf a = ldA(qb, mt * 16 + lr, 128, ks * 32, hi);
      v16bf b = ldB(kb, nt * 16 + lr, 128, ks * 32, hi);
      acc = wmma_bf16(a, b, acc);
    }
    int n = nt * 16 + lr;
    #pragma unroll
    for (int r = 0; r < 8; ++r)
      sc[(mt * 16 + r + hi * 8) * E_PAD + n] = acc[r];
  }
  __syncthreads();

  // softmax (mask cols >= E_TOT) -> bf16 attn, K padded to 128 with zeros
  if (tid < E_PAD) {
    const float scale = 0.0883883476483184f; // 1/sqrt(128)
    float mx = -3.0e38f;
    for (int c = 0; c < E_TOT; ++c) mx = fmaxf(mx, sc[tid * E_PAD + c] * scale);
    float s = 0.f;
    for (int c = 0; c < E_TOT; ++c) s += __expf(sc[tid * E_PAD + c] * scale - mx);
    float inv = 1.f / s;
    for (int c = 0; c < E_TOT; ++c)
      at[tid * 128 + c] = tobf(__expf(sc[tid * E_PAD + c] * scale - mx) * inv);
    for (int c = E_TOT; c < 128; ++c) at[tid * 128 + c] = tobf(0.f);
  }
  __syncthreads();

  // out = attn @ v   (B = v^T: [n=d][k=f])
  for (int tile = wave; tile < 7 * 8; tile += 8) {
    int mt = tile >> 3, nt = tile & 7;
    v8f acc = ZERO8;
    #pragma unroll
    for (int ks = 0; ks < 4; ++ks) {
      v16bf a = ldA(at, mt * 16 + lr, 128, ks * 32, hi);
      v16bf b = ldB(vt, nt * 16 + lr, 128, ks * 32, hi);
      acc = wmma_bf16(a, b, acc);
    }
    int n = nt * 16 + lr;
    #pragma unroll
    for (int r = 0; r < 8; ++r) {
      int m = mt * 16 + r + hi * 8;
      if (m < E_TOT) obuf[((size_t)t * E_TOT + m) * DMODEL + n] = acc[r];
    }
  }
}

// =====================================================================
// Per-agent MLP head: tanh(o @ W1 + b1) @ W2 + b2. One block per (a, t).
// =====================================================================
__global__ __launch_bounds__(256)
void mlp_kernel(const float* __restrict__ obuf,
                const float* __restrict__ W1, const float* __restrict__ B1,
                const float* __restrict__ W2, const float* __restrict__ B2,
                float* __restrict__ out)
{
  __shared__ __bf16 ob [E_PAD * 128];
  __shared__ __bf16 W1t[128 * 128];
  __shared__ __bf16 hb [E_PAD * 128];
  __shared__ __bf16 W2t[64 * 128];
  __shared__ float  b1s[128];
  __shared__ float  b2s[64];

  const int tid  = threadIdx.x;
  const int a    = blockIdx.x / T_STEPS;
  const int t    = blockIdx.x % T_STEPS;
  const int wave = tid >> 5, lane = tid & 31, lr = lane & 15, hi = lane >> 4;

  for (int i = tid; i < E_PAD * 128; i += 256) {
    int e = i >> 7, d = i & 127;
    ob[i] = tobf((e < E_TOT) ? obuf[((size_t)t * E_TOT + e) * DMODEL + d] : 0.f);
  }
  const float* W1a = W1 + (size_t)a * 128 * 128;
  for (int i = tid; i < 128 * 128; i += 256) {
    int n = i >> 7, k = i & 127;
    W1t[n * 128 + k] = tobf(W1a[k * 128 + n]);
  }
  if (tid < 128) b1s[tid] = B1[a * 128 + tid];
  const float* W2a = W2 + (size_t)a * 128 * 60;
  for (int i = tid; i < 64 * 128; i += 256) {
    int n = i >> 7, k = i & 127;
    W2t[n * 128 + k] = tobf((n < 60) ? W2a[k * 60 + n] : 0.f);
  }
  if (tid < 64) b2s[tid] = (tid < 60) ? B2[a * 60 + tid] : 0.f;
  __syncthreads();

  for (int tile = wave; tile < 7 * 8; tile += 8) {
    int mt = tile >> 3, nt = tile & 7;
    v8f acc = ZERO8;
    #pragma unroll
    for (int ks = 0; ks < 4; ++ks) {
      v16bf avv = ldA(ob, mt * 16 + lr, 128, ks * 32, hi);
      v16bf bvv = ldB(W1t, nt * 16 + lr, 128, ks * 32, hi);
      acc = wmma_bf16(avv, bvv, acc);
    }
    int n = nt * 16 + lr;
    #pragma unroll
    for (int r = 0; r < 8; ++r)
      hb[(mt * 16 + r + hi * 8) * 128 + n] = tobf(tanhf(acc[r] + b1s[n]));
  }
  __syncthreads();

  for (int tile = wave; tile < 7 * 4; tile += 8) {
    int mt = tile >> 2, nt = tile & 3;
    v8f acc = ZERO8;
    #pragma unroll
    for (int ks = 0; ks < 4; ++ks) {
      v16bf avv = ldA(hb, mt * 16 + lr, 128, ks * 32, hi);
      v16bf bvv = ldB(W2t, nt * 16 + lr, 128, ks * 32, hi);
      acc = wmma_bf16(avv, bvv, acc);
    }
    int n = nt * 16 + lr;
    #pragma unroll
    for (int r = 0; r < 8; ++r) {
      int m = mt * 16 + r + hi * 8;
      if (m < E_TOT && n < 60)
        out[(((size_t)a * T_STEPS + t) * E_TOT + m) * 60 + n] = acc[r] + b2s[n];
    }
  }
}

extern "C" void kernel_launch(void* const* d_in, const int* in_sizes, int n_in,
                              void* d_out, int out_size, void* d_ws, size_t ws_size,
                              hipStream_t stream) {
  (void)in_sizes; (void)n_in; (void)out_size; (void)ws_size;
  const float* agent_feature = (const float*)d_in[2];
  const float* map_feature   = (const float*)d_in[3];
  const float* obj_feature   = (const float*)d_in[6];
  const float* aW0=(const float*)d_in[7],  *aB0=(const float*)d_in[8];
  const float* aW1=(const float*)d_in[9],  *aB1=(const float*)d_in[10];
  const float* aW2=(const float*)d_in[11], *aB2=(const float*)d_in[12];
  const float* mW0=(const float*)d_in[13], *mB0=(const float*)d_in[14];
  const float* mW1=(const float*)d_in[15], *mB1=(const float*)d_in[16];
  const float* mW2=(const float*)d_in[17], *mB2=(const float*)d_in[18];
  const float* oW0=(const float*)d_in[19], *oB0=(const float*)d_in[20];
  const float* oW1=(const float*)d_in[21], *oB1=(const float*)d_in[22];
  const float* oW2=(const float*)d_in[23], *oB2=(const float*)d_in[24];
  const float* Wq=(const float*)d_in[25], *Wk=(const float*)d_in[26], *Wv=(const float*)d_in[27];
  const float* mlpW1=(const float*)d_in[28], *mlpB1=(const float*)d_in[29];
  const float* mlpW2=(const float*)d_in[30], *mlpB2=(const float*)d_in[31];

  float* Gbuf = (float*)d_ws;                          // [T][E_TOT][128]
  float* obuf = Gbuf + (size_t)T_STEPS * E_TOT * DMODEL;

  subnet_kernel<256><<<4  * T_STEPS, 256, 0, stream>>>(agent_feature, aW0,aB0,aW1,aB1,aW2,aB2, Gbuf, 0,  1);
  subnet_kernel<256><<<64 * T_STEPS, 256, 0, stream>>>(map_feature,   mW0,mB0,mW1,mB1,mW2,mB2, Gbuf, 4,  0);
  subnet_kernel<128><<<32 * T_STEPS, 256, 0, stream>>>(obj_feature,   oW0,oB0,oW1,oB1,oW2,oB2, Gbuf, 68, 0);
  attn_kernel<<<T_STEPS, 256, 0, stream>>>(Gbuf, Wq, Wk, Wv, obuf);
  mlp_kernel<<<4 * T_STEPS, 256, 0, stream>>>(obuf, mlpW1, mlpB1, mlpW2, mlpB2, (float*)d_out);
}
